// SwinTransformer_76888504533469
// MI455X (gfx1250) — compile-verified, matrix operand order checked by reading, not run on hardware
//
#include <hip/hip_runtime.h>
#include <hip/hip_bf16.h>

typedef __attribute__((ext_vector_type(16))) _Float16 v16h;
typedef __attribute__((ext_vector_type(8)))  _Float16 v8h;
typedef __attribute__((ext_vector_type(8)))  float    v8f;

#define BB     2
#define CC     96
#define NHEAD  4
#define HDIM   24
#define HH     186
#define WWID   360
#define WH_    6
#define WW_    12
#define NTOK   72
#define NWH_   31
#define NWW_   30
#define NWIN   930
#define SH_    3
#define SW_    6
#define SCALE_F 0.2041241452319315f  /* 24^-0.5 */

// ---------------- WMMA fragment helpers (CDNA5 16-bit A/B layout) ----------------
// A (and B^T-style) fragment: row-contiguous. lane&15 = row, lanes>=16 get K+8 offset.
// Per lane: halfs [kb..kb+7] and [kb+16..kb+23], kb = (lane>>4)*8.
__device__ __forceinline__ v16h frag_row(const _Float16* base, int ld, int lane) {
  const _Float16* p = base + (lane & 15) * ld + ((lane >> 4) << 3);
  v8h lo = *(const v8h*)p;
  v8h hi = *(const v8h*)(p + 16);
  v16h f;
#pragma unroll
  for (int e = 0; e < 8; ++e) { f[e] = lo[e]; f[8 + e] = hi[e]; }
  return f;
}

// True-B fragment from row-major [K x N]: lane&15 = column n, strided K reads.
__device__ __forceinline__ v16h frag_col(const _Float16* base, int ld, int lane) {
  const _Float16* p = base + (lane & 15) + (((lane >> 4) << 3)) * ld;
  v16h f;
#pragma unroll
  for (int e = 0; e < 8; ++e) f[e] = p[e * ld];
#pragma unroll
  for (int e = 0; e < 8; ++e) f[8 + e] = p[(16 + e) * ld];
  return f;
}

__device__ __forceinline__ v8f wmma32(v16h a, v16h b, v8f c) {
  return __builtin_amdgcn_wmma_f32_16x16x32_f16(false, a, false, b, (short)0, c, false, false);
}

__device__ __forceinline__ int regid(int h, int w) {
  int hs = (h < HH - WH_) ? 0 : ((h < HH - SH_) ? 1 : 2);
  int ws = (w < WWID - WW_) ? 0 : ((w < WWID - SW_) ? 1 : 2);
  return hs * 3 + ws;
}

__device__ __forceinline__ float gelu_f(float x) {
  return 0.5f * x * (1.0f + tanhf(0.7978845608028654f * (x + 0.044715f * x * x * x)));
}

// ---------------- one-time data movement kernels ----------------
__global__ void convert_weights_kernel(const float* __restrict__ qkv_w,
                                       const float* __restrict__ proj_w,
                                       const float* __restrict__ fc1_w,
                                       const float* __restrict__ fc2_w,
                                       _Float16* __restrict__ wbuf) {
  int i = blockIdx.x * 256 + threadIdx.x;
  if (i < 55296)       wbuf[i] = (_Float16)qkv_w[i];
  else if (i < 73728)  wbuf[i] = (_Float16)proj_w[i - 55296];
  else if (i < 147456) wbuf[i] = (_Float16)fc1_w[i - 73728];
  else if (i < 221184) wbuf[i] = (_Float16)fc2_w[i - 147456];
}

__global__ void pad_kernel(const float* __restrict__ x, float* __restrict__ xbuf) {
  long i = (long)blockIdx.x * 256 + threadIdx.x;
  if (i >= (long)BB * HH * WWID * CC) return;
  int c = (int)(i % CC); long r = i / CC;
  int w = (int)(r % WWID); r /= WWID;
  int h = (int)(r % HH);   int b = (int)(r / HH);
  float v = 0.0f;
  int lat = h - 2;
  if (lat >= 0 && lat < 181) v = x[(((long)b * CC + c) * 181 + lat) * 360 + w];
  xbuf[i] = v;
}

__global__ void out_kernel(const float* __restrict__ xbuf, float* __restrict__ out) {
  long i = (long)blockIdx.x * 256 + threadIdx.x;
  if (i >= (long)BB * CC * 181 * 360) return;
  int w = (int)(i % 360); long r = i / 360;
  int lat = (int)(r % 181); r /= 181;
  int c = (int)(r % CC);    int b = (int)(r / CC);
  out[i] = xbuf[(((long)b * HH + (lat + 2)) * WWID + w) * CC + c];
}

// ---------------- fused window attention (LN1 + QKV + attn + proj + residual) ----------------
// smem carve (halfs): xln[80*104] | qkv[3*4*96*32] | att[4*80*96]
#define XLD   104
#define XOFF  0
#define QOFF  (80 * XLD)                 // 8320
#define AOFF  (QOFF + 3 * 4 * 96 * 32)   // 45184
#define SMEMH (AOFF + 4 * 80 * 96)       // 75904 halfs = 151808 B

__global__ __launch_bounds__(128) void attn_kernel(
    const float* __restrict__ g1, const float* __restrict__ b1,
    const _Float16* __restrict__ Wq, const float* __restrict__ qkvb,
    const _Float16* __restrict__ Wp, const float* __restrict__ projb,
    const float* __restrict__ rpb, float* __restrict__ xbuf, int shift) {
  __shared__ __align__(16) _Float16 smem[SMEMH];
  _Float16* xln  = smem + XOFF;   // 80 x 104 (LN out; later reused as attn-out)
  _Float16* qkvs = smem + QOFF;   // [3][4][96][32]
  _Float16* att  = smem + AOFF;   // [4][80][96]

  const int tid  = threadIdx.x;
  const int lane = tid & 31;
  const int wv   = tid >> 5;
  const int bb   = blockIdx.x / NWIN;
  const int win  = blockIdx.x % NWIN;
  const int wh   = win / NWW_;
  const int ww   = win % NWW_;

  // zero LDS (pad rows/cols must be exact zeros to avoid NaN through WMMA)
  {
    int4* p = (int4*)smem;
    const int n16 = (SMEMH * 2) / 16;
    for (int i = tid; i < n16; i += 128) p[i] = make_int4(0, 0, 0, 0);
  }
  __syncthreads();

  // ---- LayerNorm1 with (optional) shifted-window gather ----
  if (tid < NTOK) {
    int r = tid;
    int gh = wh * WH_ + r / WW_, gw = ww * WW_ + r % WW_;
    int sh = shift ? (gh + SH_) % HH : gh;
    int sw = shift ? (gw + SW_) % WWID : gw;
    const float* px = xbuf + (((long)bb * HH + sh) * WWID + sw) * CC;
    float mean = 0.0f, sq = 0.0f;
    for (int c = 0; c < CC; ++c) { float v = px[c]; mean += v; sq += v * v; }
    mean *= (1.0f / CC);
    float var = sq * (1.0f / CC) - mean * mean;
    float rstd = rsqrtf(var + 1e-5f);
    for (int c = 0; c < CC; ++c)
      xln[r * XLD + c] = (_Float16)((px[c] - mean) * rstd * g1[c] + b1[c]);
  }
  __syncthreads();

  // ---- QKV GEMM: (80x96) @ (96x288) ; q pre-scaled; scatter into [which][head][row][dim] ----
  for (int t = wv; t < 5 * 18; t += 4) {
    int mt = t / 18, nt = t % 18;
    v8f acc = {};
#pragma unroll
    for (int kt = 0; kt < 3; ++kt) {
      v16h a  = frag_row(xln + mt * 16 * XLD + kt * 32, XLD, lane);
      v16h bf = frag_row(Wq + (nt * 16) * CC + kt * 32, CC, lane);  // B = Wq^T
      acc = wmma32(a, bf, acc);
    }
    int o = nt * 16 + (lane & 15);
    int which = o / CC, hh = (o % CC) / HDIM, dd = o % HDIM;
    float bias = qkvb[o];
    float scl = (which == 0) ? SCALE_F : 1.0f;
    int mb = mt * 16 + ((lane >> 4) << 3);
    _Float16* dst = qkvs + ((which * 4 + hh) * 96 + mb) * 32 + dd;
#pragma unroll
    for (int i = 0; i < 8; ++i) dst[i * 32] = (_Float16)((acc[i] + bias) * scl);
  }
  __syncthreads();

  // ---- QK^T per head (one head per wave), + rel-pos bias + shift mask ----
  {
    int hh = wv;
    const _Float16* qh = qkvs + ((0 * 4 + hh) * 96) * 32;
    const _Float16* kh = qkvs + ((1 * 4 + hh) * 96) * 32;
    for (int t = 0; t < 25; ++t) {
      int mt = t / 5, nt = t % 5;
      v16h a  = frag_row(qh + mt * 16 * 32, 32, lane);
      v16h bf = frag_row(kh + nt * 16 * 32, 32, lane);  // B = K^T
      v8f acc = {};
      acc = wmma32(a, bf, acc);
      int n = nt * 16 + (lane & 15);
#pragma unroll
      for (int i = 0; i < 8; ++i) {
        int m = mt * 16 + i + ((lane >> 4) << 3);
        float v = acc[i];
        if (m < NTOK && n < NTOK) {
          int ih = m / WW_, iw = m % WW_, jh = n / WW_, jw = n % WW_;
          int idx = (ih - jh + WH_ - 1) * (2 * WW_ - 1) + (iw - jw + WW_ - 1);
          v += rpb[idx * NHEAD + hh];
          if (shift) {
            int ri = regid(wh * WH_ + ih, ww * WW_ + iw);
            int rj = regid(wh * WH_ + jh, ww * WW_ + jw);
            if (ri != rj) v -= 100.0f;
          }
        }
        att[(hh * 80 + m) * 96 + n] = (_Float16)v;
      }
    }
  }
  __syncthreads();

  // ---- softmax over keys (f32 math, f16 store; zero the K-pad lanes) ----
  {
    int hh = wv;
    for (int r = lane; r < 80; r += 32) {
      _Float16* row = att + (hh * 80 + r) * 96;
      if (r < NTOK) {
        float mx = -1e30f;
        for (int j = 0; j < NTOK; ++j) mx = fmaxf(mx, (float)row[j]);
        float s = 0.0f;
        for (int j = 0; j < NTOK; ++j) s += __expf((float)row[j] - mx);
        float inv = 1.0f / s;
        for (int j = 0; j < NTOK; ++j) row[j] = (_Float16)(__expf((float)row[j] - mx) * inv);
        for (int j = NTOK; j < 96; ++j) row[j] = (_Float16)0.0f;
      } else {
        for (int j = 0; j < 96; ++j) row[j] = (_Float16)0.0f;
      }
    }
  }
  __syncthreads();

  // ---- attn @ V (per head) -> attn-out in xln (heads concat along channels) ----
  {
    int hh = wv;
    const _Float16* vh = qkvs + ((2 * 4 + hh) * 96) * 32;
    for (int t = 0; t < 10; ++t) {
      int mt = t / 2, nt = t % 2;
      v8f acc = {};
#pragma unroll
      for (int kt = 0; kt < 3; ++kt) {
        v16h a  = frag_row(att + (hh * 80 + mt * 16) * 96 + kt * 32, 96, lane);
        v16h bf = frag_col(vh + kt * 32 * 32 + nt * 16, 32, lane);  // B = V
        acc = wmma32(a, bf, acc);
      }
      int n = nt * 16 + (lane & 15);
      if (n < HDIM) {
#pragma unroll
        for (int i = 0; i < 8; ++i) {
          int m = mt * 16 + i + ((lane >> 4) << 3);
          xln[m * XLD + hh * HDIM + n] = (_Float16)acc[i];
        }
      }
    }
  }
  __syncthreads();

  // ---- proj GEMM + residual write-back (window reverse + un-shift implicit) ----
  for (int t = wv; t < 5 * 6; t += 4) {
    int mt = t / 6, nt = t % 6;
    v8f acc = {};
#pragma unroll
    for (int kt = 0; kt < 3; ++kt) {
      v16h a  = frag_row(xln + mt * 16 * XLD + kt * 32, XLD, lane);
      v16h bf = frag_row(Wp + (nt * 16) * CC + kt * 32, CC, lane);  // B = Wp^T
      acc = wmma32(a, bf, acc);
    }
    int n = nt * 16 + (lane & 15);
    float bias = projb[n];
#pragma unroll
    for (int i = 0; i < 8; ++i) {
      int m = mt * 16 + i + ((lane >> 4) << 3);
      if (m < NTOK) {
        int gh = wh * WH_ + m / WW_, gw = ww * WW_ + m % WW_;
        int sh = shift ? (gh + SH_) % HH : gh;
        int sw = shift ? (gw + SW_) % WWID : gw;
        float* px = xbuf + (((long)bb * HH + sh) * WWID + sw) * CC + n;
        *px = *px + acc[i] + bias;
      }
    }
  }
}

// ---------------- fused MLP (LN2 + FC1 + gelu + FC2 + residual), 80 pixels/WG ----------------
#define H1LD 392
__global__ __launch_bounds__(128) void mlp_kernel(
    const float* __restrict__ g2, const float* __restrict__ b2,
    const _Float16* __restrict__ W1, const float* __restrict__ fc1b,
    const _Float16* __restrict__ W2, const float* __restrict__ fc2b,
    float* __restrict__ xbuf) {
  __shared__ __align__(16) _Float16 xl[80 * XLD];
  __shared__ __align__(16) _Float16 h1[80 * H1LD];
  const int tid = threadIdx.x;
  const int lane = tid & 31;
  const int wv = tid >> 5;
  const long base = (long)blockIdx.x * 80;

  if (tid < 80) {
    const float* px = xbuf + (base + tid) * CC;
    float mean = 0.0f, sq = 0.0f;
    for (int c = 0; c < CC; ++c) { float v = px[c]; mean += v; sq += v * v; }
    mean *= (1.0f / CC);
    float var = sq * (1.0f / CC) - mean * mean;
    float rstd = rsqrtf(var + 1e-5f);
    for (int c = 0; c < CC; ++c)
      xl[tid * XLD + c] = (_Float16)((px[c] - mean) * rstd * g2[c] + b2[c]);
  }
  __syncthreads();

  // FC1: (80x96)@(96x384) + gelu
  for (int t = wv; t < 5 * 24; t += 4) {
    int mt = t / 24, nt = t % 24;
    v8f acc = {};
#pragma unroll
    for (int kt = 0; kt < 3; ++kt) {
      v16h a  = frag_row(xl + mt * 16 * XLD + kt * 32, XLD, lane);
      v16h bf = frag_row(W1 + (nt * 16) * CC + kt * 32, CC, lane);  // B = W1^T
      acc = wmma32(a, bf, acc);
    }
    int n = nt * 16 + (lane & 15);
    float bias = fc1b[n];
    int mb = mt * 16 + ((lane >> 4) << 3);
#pragma unroll
    for (int i = 0; i < 8; ++i)
      h1[(mb + i) * H1LD + n] = (_Float16)gelu_f(acc[i] + bias);
  }
  __syncthreads();

  // FC2: (80x384)@(384x96) + residual
  for (int t = wv; t < 5 * 6; t += 4) {
    int mt = t / 6, nt = t % 6;
    v8f acc = {};
#pragma unroll
    for (int kt = 0; kt < 12; ++kt) {
      v16h a  = frag_row(h1 + mt * 16 * H1LD + kt * 32, H1LD, lane);
      v16h bf = frag_row(W2 + (nt * 16) * 384 + kt * 32, 384, lane);  // B = W2^T
      acc = wmma32(a, bf, acc);
    }
    int n = nt * 16 + (lane & 15);
    float bias = fc2b[n];
    int mb = mt * 16 + ((lane >> 4) << 3);
#pragma unroll
    for (int i = 0; i < 8; ++i) {
      float* px = xbuf + (base + mb + i) * CC + n;
      *px = *px + acc[i] + bias;
    }
  }
}

// ---------------- host launcher ----------------
extern "C" void kernel_launch(void* const* d_in, const int* in_sizes, int n_in,
                              void* d_out, int out_size, void* d_ws, size_t ws_size,
                              hipStream_t stream) {
  const float* x      = (const float*)d_in[0];
  const float* g1     = (const float*)d_in[1];
  const float* b1     = (const float*)d_in[2];
  const float* qkv_w  = (const float*)d_in[3];
  const float* qkv_b  = (const float*)d_in[4];
  const float* proj_w = (const float*)d_in[5];
  const float* proj_b = (const float*)d_in[6];
  const float* rpb    = (const float*)d_in[7];
  const float* g2     = (const float*)d_in[8];
  const float* b2     = (const float*)d_in[9];
  const float* fc1_w  = (const float*)d_in[10];
  const float* fc1_b  = (const float*)d_in[11];
  const float* fc2_w  = (const float*)d_in[12];
  const float* fc2_b  = (const float*)d_in[13];
  float* out = (float*)d_out;

  _Float16* wbuf = (_Float16*)d_ws;              // 221184 f16 = 442368 B
  float* xbuf = (float*)((char*)d_ws + 442368);  // B*H*W*C f32 = 51.4 MB

  const _Float16* Wq = wbuf;                 // (2,288,96)
  const _Float16* Wp = wbuf + 55296;         // (2,96,96)
  const _Float16* W1 = wbuf + 73728;         // (2,384,96)
  const _Float16* W2 = wbuf + 147456;        // (2,96,384)

  convert_weights_kernel<<<(221184 + 255) / 256, 256, 0, stream>>>(qkv_w, proj_w, fc1_w, fc2_w, wbuf);

  const long padN = (long)BB * HH * WWID * CC;  // 12,856,320
  pad_kernel<<<(int)((padN + 255) / 256), 256, 0, stream>>>(x, xbuf);

  for (int d = 0; d < 2; ++d) {
    attn_kernel<<<BB * NWIN, 128, 0, stream>>>(
        g1 + d * CC, b1 + d * CC,
        Wq + d * 288 * CC, qkv_b + d * 288,
        Wp + d * CC * CC, proj_b + d * CC,
        rpb + d * 253 * NHEAD, xbuf, d & 1);
    mlp_kernel<<<(BB * HH * WWID) / 80, 128, 0, stream>>>(
        g2 + d * CC, b2 + d * CC,
        W1 + d * 384 * CC, fc1_b + d * 384,
        W2 + d * CC * 384, fc2_b + d * CC, xbuf);
  }

  const long outN = (long)BB * CC * 181 * 360;  // 12,510,720
  out_kernel<<<(int)((outN + 255) / 256), 256, 0, stream>>>(xbuf, out);
}
